// MultiTaskLoss_8306466750557
// MI455X (gfx1250) — matrix operand order repaired and checked by reading, not compiled, not run on hardware
//
#include <hip/hip_runtime.h>
#include <hip/hip_bf16.h>

#define BN     4096
#define KN     8192
#define NTOT   (BN + KN)          // 12288
#define DIM    128
#define NCLS   1000
#define NSPLIT 4                         // column-sweep splits (occupancy)
#define INV_TEMP 14.285714285714286f     // 1/0.07
#define C2       20.609929155556624f     // log2(e)/0.07
#define EXPM     6.248793e-7f            // 2^(-C2) = e^(-1/0.07)

typedef __attribute__((ext_vector_type(16))) __bf16 v16bf;
typedef __attribute__((ext_vector_type(8)))  __bf16 v8bf;
typedef __attribute__((ext_vector_type(8)))  float  v8f;

// ---- workspace layout (bytes) ------------------------------------------------
#define WS_FB   0                          // bf16 features: NTOT*DIM*2 = 3,145,728
#define WS_G    3145728                    // class sums:    NCLS*DIM*4 =   512,000
#define WS_CNT  (WS_G + 512000)            // class counts:  NCLS*4
#define WS_ROW  (WS_CNT + 4096)            // per-row sumexp: BN*4

// -----------------------------------------------------------------------------
// Zero G, cnt, rowsum and the scalar output (atomic accumulation targets).
__global__ __launch_bounds__(256) void init_kernel(
    float* G, int* cnt, float* rowsum, float* out) {
    const int idx = blockIdx.x * 256 + threadIdx.x;
    if (idx < NCLS * DIM) G[idx] = 0.0f;
    if (idx < NCLS)       cnt[idx] = 0;
    if (idx < BN)         rowsum[idx] = 0.0f;
    if (idx == 0)         out[0] = 0.0f;
}

// -----------------------------------------------------------------------------
// fp32 -> bf16 convert + scatter-add class sums G[lab] += F_row (fp32) + counts.
__global__ __launch_bounds__(256) void cvt_scatter_kernel(
    const float* __restrict__ in, const int* __restrict__ labels,
    __bf16* __restrict__ fb, float* __restrict__ G, int* __restrict__ cnt) {
    const int base = (blockIdx.x * 256 + threadIdx.x) * 4;   // 4 elems, same row
    if (base >= NTOT * DIM) return;
    const int row = base >> 7;            // /DIM
    const int col = base & (DIM - 1);
    const int lab = labels[row];
    const float4 f = *(const float4*)(in + base);
    fb[base + 0] = (__bf16)f.x;
    fb[base + 1] = (__bf16)f.y;
    fb[base + 2] = (__bf16)f.z;
    fb[base + 3] = (__bf16)f.w;
    float* g = G + lab * DIM + col;
    atomicAdd(g + 0, f.x);
    atomicAdd(g + 1, f.y);
    atomicAdd(g + 2, f.z);
    atomicAdd(g + 3, f.w);
    if (col == 0) atomicAdd(&cnt[lab], 1);
}

// -----------------------------------------------------------------------------
// WMMA GEMM + fused sum-of-exp.  Block (x) = 16 rows; block (y) = 1/NSPLIT of
// the 12288-column sweep.  8 waves, 32 columns (2 C tiles, 8 WMMAs) per
// wave-iteration.  Shift folded: exp((d-1)/T) = 2^(d*C2) * 2^(-C2); the
// 2^(-C2) factor is applied at finalize.  Raw v_exp_f32 is safe: |d*C2| <= 21.
__global__ __launch_bounds__(256) void supcon_gemm_kernel(
    const __bf16* __restrict__ fb, float* __restrict__ rowsum) {
    __shared__ float red[8][16];

    const int lane  = threadIdx.x & 31;
    const int wave  = threadIdx.x >> 5;
    const int l15   = lane & 15;
    const int half  = lane >> 4;
    const int mBase = blockIdx.x * 16;
    const int itLo  = blockIdx.y * (NTOT / 256 / NSPLIT);    // 12 iters per split
    const int itHi  = itLo + (NTOT / 256 / NSPLIT);

    // A fragments (16 rows x K=128), contiguous k order (same as B): lanes 0-15 hold K kb+0..15
    // contiguous, kb=32c; lanes 16-31: kb=32c+16.
    const __bf16* arow = fb + (size_t)(mBase + l15) * DIM;
    v16bf afrag[4];
#pragma unroll
    for (int c = 0; c < 4; ++c) {
        const int kb = c * 32 + half * 16;
        v8bf lo = *(const v8bf*)(arow + kb);
        v8bf hi = *(const v8bf*)(arow + kb + 8);
        afrag[c] = __builtin_shufflevector(lo, hi,
            0, 1, 2, 3, 4, 5, 6, 7, 8, 9, 10, 11, 12, 13, 14, 15);
    }

    float sumexp[8];
#pragma unroll
    for (int v = 0; v < 8; ++v) sumexp[v] = 0.0f;

    for (int it = itLo; it < itHi; ++it) {
        const int cb = (it * 8 + wave) * 32;         // 32-column block
        // Wave-uniform scalar copy so diag checks become s_cmp/s_cbranch.
        const int cbs = __builtin_amdgcn_readfirstlane(cb);
        // B uses A's contiguous k order: lanes 0-15 hold K=32c+0..15 contiguous; lanes 16-31
        // hold K=32c+16..31.  B[k][n] = F[n][k].
        const __bf16* brow0 = fb + (size_t)(cb + l15) * DIM;
        const __bf16* brow1 = brow0 + 16 * DIM;

        v8f acc0 = {}, acc1 = {};
#pragma unroll
        for (int c = 0; c < 4; ++c) {
            const int kb = c * 32 + half * 16;
            v8bf lo0 = *(const v8bf*)(brow0 + kb);
            v8bf hi0 = *(const v8bf*)(brow0 + kb + 8);
            v8bf lo1 = *(const v8bf*)(brow1 + kb);
            v8bf hi1 = *(const v8bf*)(brow1 + kb + 8);
            v16bf b0 = __builtin_shufflevector(lo0, hi0,
                0, 1, 2, 3, 4, 5, 6, 7, 8, 9, 10, 11, 12, 13, 14, 15);
            v16bf b1 = __builtin_shufflevector(lo1, hi1,
                0, 1, 2, 3, 4, 5, 6, 7, 8, 9, 10, 11, 12, 13, 14, 15);
            acc0 = __builtin_amdgcn_wmma_f32_16x16x32_bf16(
                false, afrag[c], false, b0, (short)0, acc0, false, false);
            acc1 = __builtin_amdgcn_wmma_f32_16x16x32_bf16(
                false, afrag[c], false, b1, (short)0, acc1, false, false);
        }

        // Epilogue: v_mul + v_exp + v_add per element (raw exp2 intrinsic).
        // The diagonal lands in exactly one 16-col tile per block; the check
        // is scalar (s_cbranch), the mask inside is a branchless cndmask.
        if (cbs != mBase) {
#pragma unroll
            for (int v = 0; v < 8; ++v)
                sumexp[v] += __builtin_amdgcn_exp2f(acc0[v] * C2);
        } else {
            const int j = cb + l15;
#pragma unroll
            for (int v = 0; v < 8; ++v) {
                const float e = __builtin_amdgcn_exp2f(acc0[v] * C2);
                const float m = (j != (mBase + v + 8 * half)) ? 1.0f : 0.0f;
                sumexp[v] += m * e;
            }
        }
        if (cbs + 16 != mBase) {
#pragma unroll
            for (int v = 0; v < 8; ++v)
                sumexp[v] += __builtin_amdgcn_exp2f(acc1[v] * C2);
        } else {
            const int j = cb + 16 + l15;
#pragma unroll
            for (int v = 0; v < 8; ++v) {
                const float e = __builtin_amdgcn_exp2f(acc1[v] * C2);
                const float m = (j != (mBase + v + 8 * half)) ? 1.0f : 0.0f;
                sumexp[v] += m * e;
            }
        }
    }

    // Reduce across the 16 lanes sharing each row (C layout: row = v+8*half).
#pragma unroll
    for (int v = 0; v < 8; ++v) {
        for (int m = 1; m < 16; m <<= 1)
            sumexp[v] += __shfl_xor(sumexp[v], m, 32);
    }
    if (l15 == 0) {
#pragma unroll
        for (int v = 0; v < 8; ++v) red[wave][v + 8 * half] = sumexp[v];
    }
    __syncthreads();
    if (threadIdx.x < 16) {
        float se = 0.0f;
#pragma unroll
        for (int w = 0; w < 8; ++w) se += red[w][threadIdx.x];
        atomicAdd(&rowsum[mBase + threadIdx.x], se);   // raw: sum of 2^(d*C2)
    }
}

// -----------------------------------------------------------------------------
// Per-row finalize: possum via <F_i, G[lab_i]> (fp32), diag excluded via |F_i|^2,
// poscnt = cnt[lab]-1; combine with sumexp; accumulate the scalar loss.
__global__ __launch_bounds__(256) void finalize_kernel(
    const float* __restrict__ F, const float* __restrict__ G,
    const int* __restrict__ cnt, const float* __restrict__ rowsum,
    const int* __restrict__ labels, const float* __restrict__ cw,
    float* __restrict__ out) {
    const int lane = threadIdx.x & 31;
    const int wave = threadIdx.x >> 5;
    const int row  = blockIdx.x * 8 + wave;
    if (row >= BN) return;
    const int lab = labels[row];
    const float4 f = *(const float4*)(F + (size_t)row * DIM + lane * 4);
    const float4 g = *(const float4*)(G + (size_t)lab * DIM + lane * 4);
    float dot  = f.x * g.x + f.y * g.y + f.z * g.z + f.w * g.w;
    float self = f.x * f.x + f.y * f.y + f.z * f.z + f.w * f.w;
    for (int m = 1; m < 32; m <<= 1) {
        dot  += __shfl_xor(dot,  m, 32);
        self += __shfl_xor(self, m, 32);
    }
    if (lane == 0) {
        const float pc     = (float)(cnt[lab] - 1);
        const float posdot = dot - self;                   // exclude diagonal
        const float se     = rowsum[row] * EXPM;           // undo folded shift
        // mean_log_prob_pos = INV_TEMP*(posdot - pc)/pc - log(se + 1e-12)
        const float mlpp = INV_TEMP * (posdot - pc) / pc - __logf(se + 1e-12f);
        atomicAdd(out, -mlpp * cw[lab] * (1.0f / (float)BN));
    }
}

// -----------------------------------------------------------------------------
// Cross-entropy: one block per row; log_softmax(sup_logits[i] + weight).
__global__ __launch_bounds__(256) void ce_kernel(
    const float* __restrict__ sup_logits, const float* __restrict__ weight,
    const int* __restrict__ labels, float* __restrict__ out) {
    __shared__ float sred[256];
    const int i = blockIdx.x;
    const int t = threadIdx.x;

    float x[4];
    float mx = -INFINITY;
#pragma unroll
    for (int u = 0; u < 4; ++u) {
        const int c = t + u * 256;
        x[u] = (c < NCLS) ? (sup_logits[(size_t)i * NCLS + c] + weight[c]) : -INFINITY;
        mx = fmaxf(mx, x[u]);
    }
    sred[t] = mx;
    __syncthreads();
    for (int s = 128; s > 0; s >>= 1) {
        if (t < s) sred[t] = fmaxf(sred[t], sred[t + s]);
        __syncthreads();
    }
    mx = sred[0];
    __syncthreads();

    float se = 0.0f;
#pragma unroll
    for (int u = 0; u < 4; ++u) {
        const int c = t + u * 256;
        se += (c < NCLS) ? __expf(x[u] - mx) : 0.0f;
    }
    sred[t] = se;
    __syncthreads();
    for (int s = 128; s > 0; s >>= 1) {
        if (t < s) sred[t] += sred[t + s];
        __syncthreads();
    }

    if (t == 0) {
        const int   lab  = labels[i];
        const float xt   = sup_logits[(size_t)i * NCLS + lab] + weight[lab];
        const float logp = (xt - mx) - __logf(sred[0]);
        atomicAdd(out, -logp * (1.0f / (float)BN));
    }
}

// -----------------------------------------------------------------------------
extern "C" void kernel_launch(void* const* d_in, const int* in_sizes, int n_in,
                              void* d_out, int out_size, void* d_ws, size_t ws_size,
                              hipStream_t stream) {
    const float* features     = (const float*)d_in[0]; // (NTOT, DIM)
    const float* sup_logits   = (const float*)d_in[1]; // (BN, NCLS)
    const float* weight       = (const float*)d_in[2]; // (1, NCLS)
    const float* class_weight = (const float*)d_in[3]; // (NCLS)
    const int*   labels       = (const int*)d_in[4];   // (NTOT)
    float* out = (float*)d_out;

    char* ws = (char*)d_ws;
    __bf16* fb     = (__bf16*)(ws + WS_FB);
    float*  G      = (float*)(ws + WS_G);
    int*    cnt    = (int*)(ws + WS_CNT);
    float*  rowsum = (float*)(ws + WS_ROW);

    init_kernel<<<(NCLS * DIM + 255) / 256, 256, 0, stream>>>(G, cnt, rowsum, out);

    const int n = NTOT * DIM;
    cvt_scatter_kernel<<<(n / 4 + 255) / 256, 256, 0, stream>>>(
        features, labels, fb, G, cnt);

    dim3 grid(BN / 16, NSPLIT);
    supcon_gemm_kernel<<<grid, 256, 0, stream>>>(fb, rowsum);

    finalize_kernel<<<BN / 8, 256, 0, stream>>>(
        features, G, cnt, rowsum, labels, class_weight, out);

    ce_kernel<<<BN, 256, 0, stream>>>(sup_logits, weight, labels, out);
}
